// JetEfficiencyNet_34196529611291
// MI455X (gfx1250) — matrix operand order repaired and checked by reading, not compiled, not run on hardware
//
#include <hip/hip_runtime.h>
#include <math.h>

// ---------------------------------------------------------------------------
// JetEfficiencyNet on MI455X (gfx1250).
//
// Input pointer layout (setup_inputs() dict order, params dict in insertion
// order, each (W,b) tuple flattened as W then b):
//   0 node_features [N,13] f32     1 flav_indices [N] i32
//   2 src [E] i32                  3 dst [E] i32
//   4 dR  [E] f32
//   5 emb [4,3]
//   6/7   e0_1  W[33,32]  b[32]    8/9   e0_2  W[32,32] b[32]
//   10/11 n0_11 W[16,24]  b[24]    12/13 n0_12 W[24,32] b[32]
//   14/15 n0_21 W[32,32]  b[32]    16/17 n0_22 W[32,32] b[32]
//   18/19 e1_1  W[161,96] b[96]    20/21 e1_2  W[96,32] b[32]
//   22/23 n1_11 W[80,56]  b[56]    24/25 n1_12 W[56,32] b[32]
//   26/27 n1_21 W[32,32]  b[32]    28/29 n1_22 W[32,32] b[32]
//   30/31 c1    W[80,64]  b[64]    32/33 c2    W[64,32] b[32]
//   34/35 c3    W[32,1]   b[1]
// ---------------------------------------------------------------------------

typedef __attribute__((ext_vector_type(16))) _Float16 v16h;
typedef __attribute__((ext_vector_type(8)))  _Float16 v8h;
typedef __attribute__((ext_vector_type(8)))  float    v8f;

// ---------------------------------------------------------------------------
// WMMA data-layout (CDNA5 v_wmma_f32_16x16x32_f16, wave32):
//  A (16x32, f16): lane L holds row M=L&15; lane group g=L>>4 holds
//    K = {c*32 + g*8 .. +7} in VGPRs 0-3 and K = {c*32+16+g*8 .. +7} in 4-7.
//  B (32x16, f16): lane L holds col N=L&15; group g holds K = g*8 + (slot&7) + (slot>>3)*16.
//  C/D (16x16, f32): VGPR r: lanes 0-15 -> M=r, lanes 16-31 -> M=8+r; N=L&15.
// Weights are pre-packed so each lane's 16 B halves are one contiguous 32B run.
// ---------------------------------------------------------------------------

__device__ inline void pack_one(const float* W, int Kin, int Nout, int kch,
                                int ntl, _Float16* dstp, int idx) {
  int total = kch * ntl * 512;
  if (idx < total) {
    int tile   = idx >> 9;
    int within = idx & 511;
    int lane = within >> 4;
    int slot = within & 15;
    int kc = tile / ntl;
    int nt = tile - kc * ntl;
    int K = kc * 32 + (lane >> 4) * 8 + (slot & 7) + (slot >> 3) * 16;
    int n = nt * 16 + (lane & 15);
    float v = (K < Kin) ? W[K * Nout + n] : 0.0f;
    dstp[idx] = (_Float16)v;
  }
}

__global__ void pack_weights_kernel(const float* __restrict__ w_e01,
                                    const float* __restrict__ w_e02,
                                    const float* __restrict__ w_e11,
                                    const float* __restrict__ w_e12,
                                    _Float16* __restrict__ packed) {
  int idx = blockIdx.x * blockDim.x + threadIdx.x;
  pack_one(w_e01,  33, 32, 2, 2, packed +     0, idx);  // 2048 halves
  pack_one(w_e02,  32, 32, 1, 2, packed +  2048, idx);  // 1024
  pack_one(w_e11, 161, 96, 6, 6, packed +  3072, idx);  // 18432
  pack_one(w_e12,  96, 32, 3, 2, packed + 21504, idx);  // 3072
}

__global__ void zero_f32_kernel(float* __restrict__ p, int n) {
  int i = blockIdx.x * blockDim.x + threadIdx.x;
  if (i < n) p[i] = 0.0f;
}

__global__ void build_nf_kernel(const float* __restrict__ nfeat,
                                const int* __restrict__ flav,
                                const float* __restrict__ emb,
                                float* __restrict__ nf32,
                                _Float16* __restrict__ nf16, int N) {
  int i = blockIdx.x * blockDim.x + threadIdx.x;
  if (i >= N) return;
  float v[16];
#pragma unroll
  for (int k = 0; k < 13; ++k) v[k] = nfeat[i * 13 + k];
  int f = flav[i];
#pragma unroll
  for (int k = 0; k < 3; ++k) v[13 + k] = emb[f * 3 + k];
#pragma unroll
  for (int k = 0; k < 16; ++k) {
    nf32[i * 16 + k] = v[k];
    nf16[i * 16 + k] = (_Float16)v[k];
  }
}

// Load an 8-element (16B) run of the per-edge input vector
// [feat_dst(F) | feat_src(F) | dR | zero-pad]. F is a multiple of 8 so runs
// never straddle segment boundaries.
template <int F>
__device__ inline v8h load_run(const _Float16* fd, const _Float16* fs,
                               _Float16 drh, int base) {
  if (base + 8 <= F) return *(const v8h*)(fd + base);
  if (base >= F && base + 8 <= 2 * F) return *(const v8h*)(fs + (base - F));
  v8h z = {};
  if (base == 2 * F) z[0] = drh;
  return z;
}

// Edge MLP: X[16 edges x (2F+1)] -> relu(W1,b1) -> H -> tanh(W2,b2) -> 32,
// atomically segment-summed into msum[dst] with native f32 fadd atomics.
template <int F, int H, int WAVES>
__global__ void __launch_bounds__(WAVES * 32)
edge_mlp_kernel(const _Float16* __restrict__ feat16,
                const int* __restrict__ src, const int* __restrict__ dst,
                const float* __restrict__ dRv,
                const _Float16* __restrict__ gW1,
                const _Float16* __restrict__ gW2,
                const float* __restrict__ b1, const float* __restrict__ b2,
                float* __restrict__ msum, int ntiles) {
  constexpr int IN   = 2 * F + 1;
  constexpr int KCH1 = (IN + 31) / 32;
  constexpr int NT1  = H / 16;
  constexpr int KCH2 = H / 32;
  constexpr int NB1  = KCH1 * NT1;   // # of B operands, MLP1
  constexpr int NB2  = KCH2 * 2;     // # of B operands, MLP2
  constexpr bool HOIST = (NB1 + NB2) <= 8;  // all weights fit in VGPRs
  constexpr int W1SZ = NB1 * 512;
  constexpr int W2SZ = NB2 * 512;

  __shared__ __attribute__((aligned(32))) _Float16 sW1[HOIST ? 16 : W1SZ];
  __shared__ __attribute__((aligned(32))) _Float16 sW2[HOIST ? 16 : W2SZ];
  __shared__ __attribute__((aligned(32))) _Float16 sHid[WAVES][16 * H];

  const int lane = threadIdx.x & 31;
  const int wid  = threadIdx.x >> 5;
  const int g    = lane >> 4;
  const int mrow = lane & 15;
  _Float16* hid  = sHid[wid];

  v16h B1r[HOIST ? NB1 : 1];
  v16h B2r[HOIST ? NB2 : 1];
  if constexpr (HOIST) {
    // Whole weight set lives in registers for the entire grid-stride loop.
#pragma unroll
    for (int i = 0; i < NB1; ++i)
      B1r[i] = *(const v16h*)(gW1 + i * 512 + lane * 16);
#pragma unroll
    for (int i = 0; i < NB2; ++i)
      B2r[i] = *(const v16h*)(gW2 + i * 512 + lane * 16);
  } else {
    for (int i = threadIdx.x; i < W1SZ; i += WAVES * 32) sW1[i] = gW1[i];
    for (int i = threadIdx.x; i < W2SZ; i += WAVES * 32) sW2[i] = gW2[i];
    __syncthreads();  // uniform: executed exactly once per thread
  }

  // Per-lane biases are loop-invariant (indexed by output column = lane&15).
  float bn1[NT1], bn2[2];
#pragma unroll
  for (int nt = 0; nt < NT1; ++nt) bn1[nt] = b1[nt * 16 + mrow];
#pragma unroll
  for (int nt = 0; nt < 2; ++nt) bn2[nt] = b2[nt * 16 + mrow];

  const int t0      = blockIdx.x * WAVES + wid;
  const int tstride = gridDim.x * WAVES;

  for (int t = t0; t < ntiles; t += tstride) {
    const int e  = t * 16 + mrow;
    const int di = dst[e];
    const int si = src[e];
    const _Float16* fd = feat16 + (long)di * F;
    const _Float16* fs = feat16 + (long)si * F;
    const _Float16 drh = (_Float16)dRv[e];

    // Prefetch next tile's gather rows (global_prefetch_b8) so the L2->L0
    // fill overlaps this tile's WMMA chain. Rows are <=160B (<=2 lines).
    const int tn = t + tstride;
    if (tn < ntiles) {
      const int en = tn * 16 + mrow;
      const char* pfd = (const char*)(feat16 + (long)dst[en] * F);
      const char* pfs = (const char*)(feat16 + (long)src[en] * F);
      __builtin_prefetch(pfd, 0, 3);
      __builtin_prefetch(pfs, 0, 3);
      if (F > 64) {
        __builtin_prefetch(pfd + 128, 0, 3);
        __builtin_prefetch(pfs + 128, 0, 3);
      }
    }

    // ---- MLP1: X @ W1 ----
    v8f acc[NT1] = {};
#pragma unroll
    for (int c = 0; c < KCH1; ++c) {
      const int base0 = c * 32 + g * 8;
      v8h lo = load_run<F>(fd, fs, drh, base0);
      v8h hi = load_run<F>(fd, fs, drh, base0 + 16);
      v16h A;
#pragma unroll
      for (int i = 0; i < 8; ++i) { A[i] = lo[i]; A[i + 8] = hi[i]; }
#pragma unroll
      for (int nt = 0; nt < NT1; ++nt) {
        v16h B;
        if constexpr (HOIST)
          B = B1r[c * NT1 + nt];
        else
          B = *(const v16h*)(sW1 + (c * NT1 + nt) * 512 + lane * 16);
        acc[nt] = __builtin_amdgcn_wmma_f32_16x16x32_f16(
            false, A, false, B, (short)0, acc[nt], false, false);
      }
    }

    // bias + relu, transpose via per-wave LDS (same-wave DS ops are in-order)
#pragma unroll
    for (int nt = 0; nt < NT1; ++nt) {
#pragma unroll
      for (int r = 0; r < 8; ++r) {
        float v = acc[nt][r] + bn1[nt];
        v = v > 0.0f ? v : 0.0f;
        hid[(r + 8 * g) * H + nt * 16 + mrow] = (_Float16)v;
      }
    }

    // ---- MLP2: hidden @ W2 ----
    v8f acc2[2] = {};
#pragma unroll
    for (int c = 0; c < KCH2; ++c) {
      const int base0 = c * 32 + g * 8;
      v8h lo = *(const v8h*)(hid + mrow * H + base0);
      v8h hi = *(const v8h*)(hid + mrow * H + base0 + 16);
      v16h A;
#pragma unroll
      for (int i = 0; i < 8; ++i) { A[i] = lo[i]; A[i + 8] = hi[i]; }
#pragma unroll
      for (int nt = 0; nt < 2; ++nt) {
        v16h B;
        if constexpr (HOIST)
          B = B2r[c * 2 + nt];
        else
          B = *(const v16h*)(sW2 + (c * 2 + nt) * 512 + lane * 16);
        acc2[nt] = __builtin_amdgcn_wmma_f32_16x16x32_f16(
            false, A, false, B, (short)0, acc2[nt], false, false);
      }
    }

    // bias + tanh + atomic segment-sum into msum[dst]
    int dn[8];
#pragma unroll
    for (int r = 0; r < 8; ++r) dn[r] = dst[t * 16 + r + 8 * g];
#pragma unroll
    for (int nt = 0; nt < 2; ++nt) {
#pragma unroll
      for (int r = 0; r < 8; ++r) {
        float v = tanhf(acc2[nt][r] + bn2[nt]);
        unsafeAtomicAdd(msum + (long)dn[r] * 32 + nt * 16 + mrow, v);
      }
    }
  }
}

// ---------------------------------------------------------------------------
// Node-side MLPs (small: VALU per-thread; weights broadcast via SGPR loads).
// ---------------------------------------------------------------------------
template <int DIN, int DOUT, int ACT>  // ACT: 0=linear 1=relu 2=tanh
__device__ inline void dense(const float* x, const float* __restrict__ W,
                             const float* __restrict__ b, float* y) {
  for (int j = 0; j < DOUT; ++j) {
    float s = b[j];
    for (int k = 0; k < DIN; ++k) s = fmaf(x[k], W[k * DOUT + j], s);
    if (ACT == 1) s = fmaxf(s, 0.0f);
    if (ACT == 2) s = tanhf(s);
    y[j] = s;
  }
}

__global__ void node0_kernel(const float* __restrict__ nf32,
                             const float* __restrict__ msum0,
                             const float* W011, const float* b011,
                             const float* W012, const float* b012,
                             const float* W021, const float* b021,
                             const float* W022, const float* b022,
                             float* __restrict__ feat1_32,
                             _Float16* __restrict__ feat1_16, int N) {
  int i = blockIdx.x * blockDim.x + threadIdx.x;
  if (i >= N) return;
  float x[16];
#pragma unroll
  for (int k = 0; k < 16; ++k) x[k] = nf32[i * 16 + k];
  float t1[24], o1[32];
  dense<16, 24, 1>(x, W011, b011, t1);
  dense<24, 32, 2>(t1, W012, b012, o1);
  float m[32];
#pragma unroll
  for (int k = 0; k < 32; ++k) m[k] = msum0[i * 32 + k];
  float t2[32], o2[32];
  dense<32, 32, 1>(m, W021, b021, t2);
  dense<32, 32, 2>(t2, W022, b022, o2);
  float ss = 0.0f;
#pragma unroll
  for (int k = 0; k < 32; ++k) ss += o1[k] * o1[k] + o2[k] * o2[k];
  float inv = rsqrtf(ss);
  float* row = feat1_32 + (long)i * 80;
  _Float16* rowh = feat1_16 + (long)i * 80;
#pragma unroll
  for (int k = 0; k < 16; ++k) { row[k] = x[k]; rowh[k] = (_Float16)x[k]; }
#pragma unroll
  for (int k = 0; k < 32; ++k) {
    float v = o1[k] * inv; row[16 + k] = v; rowh[16 + k] = (_Float16)v;
  }
#pragma unroll
  for (int k = 0; k < 32; ++k) {
    float v = o2[k] * inv; row[48 + k] = v; rowh[48 + k] = (_Float16)v;
  }
}

__global__ void node1_kernel(const float* __restrict__ feat1_32,
                             const float* __restrict__ msum1,
                             const float* W111, const float* b111,
                             const float* W112, const float* b112,
                             const float* W121, const float* b121,
                             const float* W122, const float* b122,
                             const float* c1W, const float* c1b,
                             const float* c2W, const float* c2b,
                             const float* c3W, const float* c3b,
                             float* __restrict__ out, int N) {
  int i = blockIdx.x * blockDim.x + threadIdx.x;
  if (i >= N) return;
  float x[80];
  for (int k = 0; k < 80; ++k) x[k] = feat1_32[(long)i * 80 + k];
  float t1[56], o1[32];
  dense<80, 56, 1>(x, W111, b111, t1);
  dense<56, 32, 2>(t1, W112, b112, o1);
  float m[32];
#pragma unroll
  for (int k = 0; k < 32; ++k) m[k] = msum1[i * 32 + k];
  float t2[32], o2[32];
  dense<32, 32, 1>(m, W121, b121, t2);
  dense<32, 32, 2>(t2, W122, b122, o2);
  float ss = 0.0f;
#pragma unroll
  for (int k = 0; k < 32; ++k) ss += o1[k] * o1[k] + o2[k] * o2[k];
  float inv = rsqrtf(ss);
  float y[80];
#pragma unroll
  for (int k = 0; k < 16; ++k) y[k] = x[k];
#pragma unroll
  for (int k = 0; k < 32; ++k) y[16 + k] = o1[k] * inv;
#pragma unroll
  for (int k = 0; k < 32; ++k) y[48 + k] = o2[k] * inv;
  float z1[64], z2[32];
  dense<80, 64, 1>(y, c1W, c1b, z1);
  dense<64, 32, 1>(z1, c2W, c2b, z2);
  float s = c3b[0];
#pragma unroll
  for (int k = 0; k < 32; ++k) s = fmaf(z2[k], c3W[k], s);
  out[i] = s;
}

// ---------------------------------------------------------------------------
extern "C" void kernel_launch(void* const* d_in, const int* in_sizes, int n_in,
                              void* d_out, int out_size, void* d_ws,
                              size_t ws_size, hipStream_t stream) {
  (void)n_in; (void)out_size; (void)ws_size;
  const float* node_features = (const float*)d_in[0];
  const int*   flav          = (const int*)d_in[1];
  const int*   src           = (const int*)d_in[2];
  const int*   dst           = (const int*)d_in[3];
  const float* dR            = (const float*)d_in[4];
  const float* emb   = (const float*)d_in[5];
  const float* e01W  = (const float*)d_in[6];
  const float* e01b  = (const float*)d_in[7];
  const float* e02W  = (const float*)d_in[8];
  const float* e02b  = (const float*)d_in[9];
  const float* n011W = (const float*)d_in[10];
  const float* n011b = (const float*)d_in[11];
  const float* n012W = (const float*)d_in[12];
  const float* n012b = (const float*)d_in[13];
  const float* n021W = (const float*)d_in[14];
  const float* n021b = (const float*)d_in[15];
  const float* n022W = (const float*)d_in[16];
  const float* n022b = (const float*)d_in[17];
  const float* e11W  = (const float*)d_in[18];
  const float* e11b  = (const float*)d_in[19];
  const float* e12W  = (const float*)d_in[20];
  const float* e12b  = (const float*)d_in[21];
  const float* n111W = (const float*)d_in[22];
  const float* n111b = (const float*)d_in[23];
  const float* n112W = (const float*)d_in[24];
  const float* n112b = (const float*)d_in[25];
  const float* n121W = (const float*)d_in[26];
  const float* n121b = (const float*)d_in[27];
  const float* n122W = (const float*)d_in[28];
  const float* n122b = (const float*)d_in[29];
  const float* c1W   = (const float*)d_in[30];
  const float* c1b   = (const float*)d_in[31];
  const float* c2W   = (const float*)d_in[32];
  const float* c2b   = (const float*)d_in[33];
  const float* c3W   = (const float*)d_in[34];
  const float* c3b   = (const float*)d_in[35];

  const int N = in_sizes[1];
  const int E = in_sizes[2];
  float* out = (float*)d_out;

  // Workspace carving (256B aligned).
  char* base = (char*)d_ws;
  size_t off = 0;
  auto carve = [&](size_t bytes) -> char* {
    char* p = base + off;
    off = (off + bytes + 255) & ~(size_t)255;
    return p;
  };
  float*     nf32     = (float*)carve((size_t)N * 16 * 4);
  _Float16*  nf16     = (_Float16*)carve((size_t)N * 16 * 2);
  float*     msum0    = (float*)carve((size_t)N * 32 * 4);
  float*     msum1    = (float*)carve((size_t)N * 32 * 4);
  float*     feat1_32 = (float*)carve((size_t)N * 80 * 4);
  _Float16*  feat1_16 = (_Float16*)carve((size_t)N * 80 * 2);
  _Float16*  packed   = (_Float16*)carve((size_t)24576 * 2);

  const int nblk = (N + 255) / 256;

  zero_f32_kernel<<<(N * 32 + 255) / 256, 256, 0, stream>>>(msum0, N * 32);
  zero_f32_kernel<<<(N * 32 + 255) / 256, 256, 0, stream>>>(msum1, N * 32);
  pack_weights_kernel<<<(18432 + 255) / 256, 256, 0, stream>>>(
      e01W, e02W, e11W, e12W, packed);
  build_nf_kernel<<<nblk, 256, 0, stream>>>(node_features, flav, emb, nf32,
                                            nf16, N);

  const int ntiles = E / 16;  // E is a multiple of 16
  edge_mlp_kernel<16, 32, 8><<<512, 256, 0, stream>>>(
      nf16, src, dst, dR, packed + 0, packed + 2048, e01b, e02b, msum0,
      ntiles);

  node0_kernel<<<nblk, 256, 0, stream>>>(nf32, msum0, n011W, n011b, n012W,
                                         n012b, n021W, n021b, n022W, n022b,
                                         feat1_32, feat1_16, N);

  edge_mlp_kernel<80, 96, 4><<<1024, 128, 0, stream>>>(
      feat1_16, src, dst, dR, packed + 3072, packed + 21504, e11b, e12b,
      msum1, ntiles);

  node1_kernel<<<nblk, 256, 0, stream>>>(feat1_32, msum1, n111W, n111b, n112W,
                                         n112b, n121W, n121b, n122W, n122b,
                                         c1W, c1b, c2W, c2b, c3W, c3b, out, N);
}
